// MyModel_87522843560197
// MI455X (gfx1250) — matrix operand order repaired and checked by reading, not compiled, not run on hardware
//
#include <hip/hip_runtime.h>

#define BATCH 256
#define SEQT  2048
#define EMB   128
#define HID   128
#define H3    384

typedef __attribute__((ext_vector_type(16))) __bf16 v16bf;
typedef __attribute__((ext_vector_type(8)))  __bf16 v8bf;
typedef __attribute__((ext_vector_type(8)))  float  v8f;

// ---- LDS layout (bytes) ----
// Wt  : [384][256] bf16 fused weights, K-major: Wt[n][k], k<128 -> W[k][n], k>=128 -> U[k-128][n]
// Ash : [16][256]  bf16 A-matrix: cols 0..127 = emb_t tile, cols 128..255 = h (bf16 copy)
// ZR  : [16][128]  float2 interleaved {z, r} gates
#define OFF_WT    0
#define OFF_ASH   196608
#define OFF_ZR    204800
#define OFF_BZR   221184
#define OFF_BX    222208
#define OFF_BU    222720
#define SMEM_BYTES 223232

__device__ __forceinline__ unsigned short f2bf(float f) {
  unsigned int u = __float_as_uint(f);
  u += 0x7FFFu + ((u >> 16) & 1u);   // round-to-nearest-even
  return (unsigned short)(u >> 16);
}

// Branch-free activations on the native TRANS pipe (v_exp_f32 is exp2):
//   sigmoid(x) = rcp(1 + exp2(-x*log2e))         (rcp(inf)=0 -> correct saturation)
//   tanh(y)    = (t-1)*rcp(t+1), t = exp2(min(2*y*log2e, 80))   (clamp kills inf*0)
#define LOG2E 1.4426950408889634f
__device__ __forceinline__ float fast_sigmoid(float x) {
  float t = __builtin_amdgcn_exp2f(x * -LOG2E);
  return __builtin_amdgcn_rcpf(1.f + t);
}
__device__ __forceinline__ float fast_tanh(float y) {
  float a = fminf(y * (2.f * LOG2E), 80.f);
  float t = __builtin_amdgcn_exp2f(a);
  return (t - 1.f) * __builtin_amdgcn_rcpf(t + 1.f);
}

// A tile (16x32 bf16): lane<16 -> row=lane, K k0..k0+15 ; lane>=16 -> row=lane-16, K k0+16..31
__device__ __forceinline__ v16bf load_a(const unsigned short* Ash, int ln, int hs, int kb) {
  const unsigned short* p = Ash + ln * 256 + kb + hs * 16;
  return *(const v16bf*)p;
}

// B tile (32x16 bf16): lane<16 -> col=lane, K k0..k0+15 ; lane>=16 -> col=lane-16, K k0+16..31
__device__ __forceinline__ v16bf load_b(const unsigned short* Wt, int ct, int ln, int hs, int kb) {
  return *(const v16bf*)(Wt + (ct * 16 + ln) * 256 + kb + hs * 16);
}

__device__ __forceinline__ v8f wmma_bf16(v16bf a, v16bf b, v8f c) {
  return __builtin_amdgcn_wmma_f32_16x16x32_bf16(false, a, false, b, (short)0, c, false, false);
}

__device__ __forceinline__ uint4 pack_bf16x8(float4 a0, float4 a1) {
  uint4 p;
  p.x = (unsigned)f2bf(a0.x) | ((unsigned)f2bf(a0.y) << 16);
  p.y = (unsigned)f2bf(a0.z) | ((unsigned)f2bf(a0.w) << 16);
  p.z = (unsigned)f2bf(a1.x) | ((unsigned)f2bf(a1.y) << 16);
  p.w = (unsigned)f2bf(a1.z) | ((unsigned)f2bf(a1.w) << 16);
  return p;
}

__global__ __launch_bounds__(384, 1)
void gru_fused_kernel(const int* __restrict__ xidx, const float* __restrict__ embt,
                      const float* __restrict__ W, const float* __restrict__ U,
                      const float* __restrict__ b, float* __restrict__ out,
                      float* __restrict__ state_out, float* __restrict__ flags) {
  extern __shared__ unsigned char smem[];
  unsigned short* Wt  = (unsigned short*)(smem + OFF_WT);
  unsigned short* Ash = (unsigned short*)(smem + OFF_ASH);
  float* zrb = (float*)(smem + OFF_ZR);   // interleaved {z,r} per (m,c)
  float* bzr = (float*)(smem + OFF_BZR);
  float* bxs = (float*)(smem + OFF_BX);
  float* bus = (float*)(smem + OFF_BU);

  const int tid  = threadIdx.x;
  const int wave = tid >> 5;        // 0..11
  const int lane = tid & 31;
  const int ln   = lane & 15;
  const int hs   = lane >> 4;       // half-wave select
  const int bm   = blockIdx.x * 16; // batch tile base

  // ---- one-time per-block init ----
  for (int e = tid; e < H3 * 256; e += 384) {        // weights -> LDS (bf16, K-major)
    int k = e / H3, n = e % H3;
    float v = (k < 128) ? W[k * H3 + n] : U[(k - 128) * H3 + n];
    Wt[n * 256 + k] = f2bf(v);
  }
  if (tid < 256) bzr[tid] = b[tid] + b[H3 + tid];    // fused bias for z,r columns
  if (tid < 128) { bxs[tid] = b[256 + tid]; bus[tid] = b[H3 + 256 + tid]; }
  for (int e = tid; e < 16 * HID; e += 384)          // h0 = 0 (bf16 copy)
    Ash[(e >> 7) * 256 + 128 + (e & 127)] = 0;
  if (tid < 256) {                                   // gather emb for t=0
    int m = tid >> 4, seg = tid & 15;
    int idx = xidx[(size_t)(bm + m) * SEQT];
    const float4* s = (const float4*)(embt + (size_t)idx * EMB + seg * 8);
    *(uint4*)(Ash + m * 256 + seg * 8) = pack_bf16x8(s[0], s[1]);
  }
  __syncthreads();                                    // Wt/bias ready for register preload

  if (wave < 8) {
    // ================= z / r gate waves =================
    const int ct0 = 2 * wave, ct1 = ct0 + 1;
    v16bf B0[8], B1[8];                               // weights live in VGPRs for all 2048 steps
    #pragma unroll
    for (int k = 0; k < 8; ++k) {
      B0[k] = load_b(Wt, ct0, ln, hs, 32 * k);
      B1[k] = load_b(Wt, ct1, ln, hs, 32 * k);
    }
    const int c0 = ct0 * 16 + ln, c1 = ct1 * 16 + ln;
    const float bb0 = bzr[c0], bb1 = bzr[c1];
    float* dst0 = zrb + ((c0 < 128) ? c0 * 2 : (c0 - 128) * 2 + 1);
    float* dst1 = zrb + ((c1 < 128) ? c1 * 2 : (c1 - 128) * 2 + 1);

    const int m16 = tid >> 4, seg = tid & 15;
    const size_t xrow = (size_t)(bm + m16) * SEQT;
    unsigned short* embdst = Ash + m16 * 256 + seg * 8;
    int idx_next = xidx[xrow + 1];                    // index for t=1 (SEQT > 1)

    for (int t = 0; t < SEQT; ++t) {
      __syncthreads();                                // Ash (emb_t + h_{t-1}) ready
      // issue next-step gather early: emb row for t+1 and x index for t+2 (independent)
      const bool have = (t + 1 < SEQT);
      float4 g0, g1;
      if (have) {
        const float4* s = (const float4*)(embt + (size_t)idx_next * EMB + seg * 8);
        g0 = s[0]; g1 = s[1];
      }
      int idx_fut = (t + 2 < SEQT) ? xidx[xrow + t + 2] : 0;

      v8f a0 = {}, a1 = {};
      #pragma unroll
      for (int k = 0; k < 8; ++k) {
        v16bf a = load_a(Ash, ln, hs, 32 * k);
        a0 = wmma_bf16(a, B0[k], a0);
        a1 = wmma_bf16(a, B1[k], a1);
      }
      #pragma unroll
      for (int i = 0; i < 8; ++i) {
        int m = i + hs * 8;
        dst0[m * 256] = fast_sigmoid(a0[i] + bb0);
        dst1[m * 256] = fast_sigmoid(a1[i] + bb1);
      }
      __syncthreads();                                // gates published; Ash reads done
      if (have) *(uint4*)embdst = pack_bf16x8(g0, g1);
      idx_next = idx_fut;
    }
  } else {
    // ================= candidate waves =================
    const int j0 = 2 * (wave - 8), j1 = j0 + 1;
    v16bf BX0[4], BX1[4], BU0[4], BU1[4];
    #pragma unroll
    for (int k = 0; k < 4; ++k) {
      BX0[k] = load_b(Wt, 16 + j0, ln, hs, 32 * k);
      BX1[k] = load_b(Wt, 16 + j1, ln, hs, 32 * k);
      BU0[k] = load_b(Wt, 16 + j0, ln, hs, 128 + 32 * k);
      BU1[k] = load_b(Wt, 16 + j1, ln, hs, 128 + 32 * k);
    }
    const int hc0 = j0 * 16 + ln, hc1 = j1 * 16 + ln;
    const float bx0 = bxs[hc0], bx1 = bxs[hc1];
    const float bu0 = bus[hc0], bu1 = bus[hc1];
    float h0[8], h1[8];                               // h state lives in VGPRs
    #pragma unroll
    for (int i = 0; i < 8; ++i) { h0[i] = 0.f; h1[i] = 0.f; }

    for (int t = 0; t < SEQT; ++t) {
      __syncthreads();
      v8f xa0 = {}, xa1 = {}, ua0 = {}, ua1 = {};
      #pragma unroll
      for (int k = 0; k < 4; ++k) {
        v16bf ae = load_a(Ash, ln, hs, 32 * k);        // emb part (K 0..127)
        v16bf ah = load_a(Ash, ln, hs, 128 + 32 * k);  // h   part (K 128..255)
        xa0 = wmma_bf16(ae, BX0[k], xa0);
        xa1 = wmma_bf16(ae, BX1[k], xa1);
        ua0 = wmma_bf16(ah, BU0[k], ua0);
        ua1 = wmma_bf16(ah, BU1[k], ua1);
      }
      __syncthreads();                                 // wait for z,r
      #pragma unroll
      for (int i = 0; i < 8; ++i) {
        int m = i + hs * 8;
        float2 g0v = *(const float2*)(zrb + (m * 128 + hc0) * 2);
        float2 g1v = *(const float2*)(zrb + (m * 128 + hc1) * 2);
        float hh0 = fast_tanh((xa0[i] + bx0) + g0v.y * (ua0[i] + bu0));
        float hh1 = fast_tanh((xa1[i] + bx1) + g1v.y * (ua1[i] + bu1));
        float hn0 = hh0 + g0v.x * (h0[i] - hh0);       // z*h + (1-z)*hh
        float hn1 = hh1 + g1v.x * (h1[i] - hh1);
        h0[i] = hn0; h1[i] = hn1;
        Ash[m * 256 + 128 + hc0] = f2bf(hn0);          // bf16 copy for next step's A
        Ash[m * 256 + 128 + hc1] = f2bf(hn1);
        size_t ob = ((size_t)(bm + m) * SEQT + t) * HID;
        out[ob + hc0] = hn0;
        out[ob + hc1] = hn1;
      }
      if (t == SEQT - 1) {                             // final state straight from registers
        #pragma unroll
        for (int i = 0; i < 8; ++i) {
          int m = i + hs * 8;
          state_out[(size_t)(bm + m) * HID + hc0] = h0[i];
          state_out[(size_t)(bm + m) * HID + hc1] = h1[i];
        }
      }
    }
  }

  if (blockIdx.x == 0 && tid == 0) { flags[0] = 1.0f; flags[1] = 0.0f; }
}

extern "C" void kernel_launch(void* const* d_in, const int* in_sizes, int n_in,
                              void* d_out, int out_size, void* d_ws, size_t ws_size,
                              hipStream_t stream) {
  (void)in_sizes; (void)n_in; (void)out_size; (void)d_ws; (void)ws_size;
  const int*   x   = (const int*)d_in[0];
  const float* emb = (const float*)d_in[1];
  const float* W   = (const float*)d_in[2];
  const float* U   = (const float*)d_in[3];
  const float* b   = (const float*)d_in[4];
  float* out   = (float*)d_out;
  float* state = out + (size_t)BATCH * SEQT * HID;
  float* flags = state + (size_t)BATCH * HID;

  (void)hipFuncSetAttribute(reinterpret_cast<const void*>(gru_fused_kernel),
                            hipFuncAttributeMaxDynamicSharedMemorySize, SMEM_BYTES);
  gru_fused_kernel<<<dim3(BATCH / 16), dim3(384), SMEM_BYTES, stream>>>(
      x, emb, W, U, b, out, state, flags);
}
